// DigitConvolutionalModel_69389491634952
// MI455X (gfx1250) — matrix operand order repaired and checked by reading, not compiled, not run on hardware
//
#include <hip/hip_runtime.h>

typedef _Float16 v16h __attribute__((ext_vector_type(16)));
typedef _Float16 v8h  __attribute__((ext_vector_type(8)));
typedef float    v8f  __attribute__((ext_vector_type(8)));

#define BATCH      65536
#define K1         676            // 26*26 conv outputs
#define K1PAD      704            // 22 * 32
#define NKC1       22             // K chunks for GEMM1
#define HID        256
#define NT1        16             // 256/16 N tiles for GEMM1
#define K2         256
#define NKC2       8              // K chunks for GEMM2
#define NOUT       10
#define WPB        4              // waves per block
#define HSTRIDE    264            // padded row stride (halves) for h in LDS
#define W1P_HALVES (NKC1 * NT1 * 32 * 16)   // 180224 halves
#define W2P_HALVES (NKC2 * 32 * 16)         // 4096 halves

// ---------------------------------------------------------------------------
// Pack W1[256,676] (row-major, torch [out,in]) into WMMA B-fragment order:
// half index = ((kc*16 + nt)*32 + lane)*16 + j
//   n = nt*16 + (lane&15);  K = kc*32 + (lane>>4)*8 + (j&7) + (j>>3)*16;  value = W1[n][K]
// (lanes 0-15 hold K 0..7,16..23 of the 32-K chunk, lanes 16-31 hold K 8..15,24..31)
// ---------------------------------------------------------------------------
__global__ void pack_w1_kernel(const float* __restrict__ W1,
                               _Float16* __restrict__ w1p) {
    int idx = blockIdx.x * blockDim.x + threadIdx.x;       // 0 .. W1P_HALVES-1
    int j    = idx & 15;
    int lane = (idx >> 4) & 31;
    int nt   = (idx >> 9) & 15;
    int kc   = idx >> 13;
    int n = nt * 16 + (lane & 15);
    int K = kc * 32 + (lane >> 4) * 8 + (j & 7) + (j >> 3) * 16;
    float v = (K < K1) ? W1[n * K1 + K] : 0.0f;
    w1p[idx] = (_Float16)v;
}

// Pack W2[10,256] the same way (single N tile, N padded 10 -> 16).
__global__ void pack_w2_kernel(const float* __restrict__ W2,
                               _Float16* __restrict__ w2p) {
    int idx = blockIdx.x * blockDim.x + threadIdx.x;       // 0 .. W2P_HALVES-1
    int j    = idx & 15;
    int lane = (idx >> 4) & 31;
    int kc   = idx >> 9;
    int n = lane & 15;
    int K = kc * 32 + (lane >> 4) * 8 + (j & 7) + (j >> 3) * 16;
    float v = (n < NOUT) ? W2[n * K2 + K] : 0.0f;
    w2p[idx] = (_Float16)v;
}

// ---------------------------------------------------------------------------
// Fused conv3x3 -> GEMM1(676->256) + bias + relu -> GEMM2(256->10) + bias.
// One wave per 16-row batch tile; conv outputs are generated lane-exactly in
// WMMA A-fragment layout (no staging, each conv pixel computed once).
// ---------------------------------------------------------------------------
__global__ void __launch_bounds__(WPB * 32)
fused_mlp_kernel(const float* __restrict__ x,
                 const float* __restrict__ cw,
                 const float* __restrict__ b1,
                 const float* __restrict__ b2,
                 const _Float16* __restrict__ w1p,
                 const _Float16* __restrict__ w2p,
                 float* __restrict__ out) {
    __shared__ __align__(16) _Float16 hsh[WPB][16][HSTRIDE];

    const int lane = threadIdx.x & 31;
    const int wave = threadIdx.x >> 5;
    const int m    = lane & 15;       // row within 16-row tile
    const int hi   = lane >> 4;       // 0: low half-wave, 1: high half-wave
    const int rowbase = (blockIdx.x * WPB + wave) * 16;

    const float* xrow = x + (long)(rowbase + m) * 784;

    // 3x3 kernel: uniform scalar loads
    const float w0 = cw[0], w1 = cw[1], w2 = cw[2];
    const float w3 = cw[3], w4 = cw[4], w5 = cw[5];
    const float w6 = cw[6], w7 = cw[7], w8 = cw[8];

    v8f acc[NT1] = {};

    const v16h* w1frag = (const v16h*)w1p;   // one v16h (32B) per lane per frag

    for (int kc = 0; kc < NKC1; ++kc) {
        // -------- build A fragment: conv outputs for this lane's K slots ----
        const int kbase = kc * 32 + hi * 8;
        v16h a;
#pragma unroll
        for (int j = 0; j < 16; ++j) {
            const int K  = kbase + j + ((j >= 8) ? 8 : 0);  // {0..7,16..23}+kbase
            const bool ok = (K < K1);
            int r = K / 26;
            int c = K - r * 26;
            int off = ok ? (r * 28 + c) : 0;                // branchless, in-bounds
            const float* p = xrow + off;
            float v = p[0]  * w0 + p[1]  * w1 + p[2]  * w2
                    + p[28] * w3 + p[29] * w4 + p[30] * w5
                    + p[56] * w6 + p[57] * w7 + p[58] * w8;
            a[j] = (_Float16)(ok ? v : 0.0f);
        }

        // prefetch next K-chunk of packed W1 (gfx1250 global_prefetch path)
        if (kc + 1 < NKC1)
            __builtin_prefetch((const void*)(w1frag + ((kc + 1) * NT1 * 32 + lane)), 0, 1);

        // -------- 16 WMMAs: all 256 hidden units for this K chunk -----------
#pragma unroll
        for (int nt = 0; nt < NT1; ++nt) {
            v16h b = w1frag[(kc * NT1 + nt) * 32 + lane];
            acc[nt] = __builtin_amdgcn_wmma_f32_16x16x32_f16(
                false, a, false, b, (short)0, acc[nt], false, false);
        }
    }

    // -------- bias + relu, deposit h (f16) into LDS in row-major ------------
    // C/D layout: VGPR v -> row (hi?8+v:v), col N = nt*16 + m
#pragma unroll
    for (int nt = 0; nt < NT1; ++nt) {
        const float bias = b1[nt * 16 + m];
#pragma unroll
        for (int v = 0; v < 8; ++v) {
            float h = acc[nt][v] + bias;
            h = h > 0.0f ? h : 0.0f;
            hsh[wave][hi * 8 + v][nt * 16 + m] = (_Float16)h;
        }
    }
    __syncthreads();

    // -------- GEMM2: [16x256] x [256x16(pad of 10)] -------------------------
    const v16h* w2frag = (const v16h*)w2p;
    v8f acc2 = {};
#pragma unroll
    for (int k2 = 0; k2 < NKC2; ++k2) {
        const int off0 = k2 * 32 + hi * 8;
        const v8h* plo = (const v8h*)&hsh[wave][m][off0];
        const v8h* phi = (const v8h*)&hsh[wave][m][off0 + 16];
        v8h lo = *plo;
        v8h hh = *phi;
        v16h a2;
#pragma unroll
        for (int i = 0; i < 8; ++i) { a2[i] = lo[i]; a2[8 + i] = hh[i]; }
        v16h b = w2frag[k2 * 32 + lane];
        acc2 = __builtin_amdgcn_wmma_f32_16x16x32_f16(
            false, a2, false, b, (short)0, acc2, false, false);
    }

    // -------- bias + store logits (only N < 10 valid) -----------------------
    const float bias2 = (m < NOUT) ? b2[m] : 0.0f;
    if (m < NOUT) {
#pragma unroll
        for (int v = 0; v < 8; ++v) {
            int row = rowbase + hi * 8 + v;
            out[(long)row * NOUT + m] = acc2[v] + bias2;
        }
    }
}

// ---------------------------------------------------------------------------
extern "C" void kernel_launch(void* const* d_in, const int* in_sizes, int n_in,
                              void* d_out, int out_size, void* d_ws, size_t ws_size,
                              hipStream_t stream) {
    const float* x   = (const float*)d_in[0];
    const float* cw  = (const float*)d_in[1];
    const float* W1  = (const float*)d_in[2];
    const float* b1  = (const float*)d_in[3];
    const float* W2  = (const float*)d_in[4];
    const float* b2  = (const float*)d_in[5];
    float* out = (float*)d_out;

    _Float16* w1p = (_Float16*)d_ws;
    _Float16* w2p = w1p + W1P_HALVES;   // 360448 B offset, 32B-aligned

    pack_w1_kernel<<<W1P_HALVES / 256, 256, 0, stream>>>(W1, w1p);
    pack_w2_kernel<<<W2P_HALVES / 256, 256, 0, stream>>>(W2, w2p);

    const int blocks = BATCH / (16 * WPB);   // 1024
    fused_mlp_kernel<<<blocks, WPB * 32, 0, stream>>>(x, cw, b1, b2, w1p, w2p, out);
}
